// Attention_33758442947414
// MI455X (gfx1250) — matrix operand order repaired and checked by reading, not compiled, not run on hardware
//
#include <hip/hip_runtime.h>

// ---------------------------------------------------------------------------
// Problem constants (match reference setup_inputs)
// ---------------------------------------------------------------------------
#define BATCH     2
#define SEQ       2048
#define HID       2048
#define N_HEADS   16
#define HEAD_DIM  128
#define NUM_BUCKETS 32

typedef __attribute__((ext_vector_type(16))) __bf16 v16bf;
typedef __attribute__((ext_vector_type(8)))  __bf16 v8bf;
typedef __attribute__((ext_vector_type(8)))  float  v8f;
typedef __attribute__((ext_vector_type(4)))  unsigned int u32x4;
typedef __attribute__((ext_vector_type(8)))  int i32x8;
typedef __attribute__((ext_vector_type(4)))  int i32x4;

__device__ __forceinline__ v8bf ld8(const __bf16* p) {
  return *(const v8bf*)p;
}
__device__ __forceinline__ v16bf cat16(v8bf lo, v8bf hi) {
  return __builtin_shufflevector(lo, hi, 0,1,2,3,4,5,6,7,8,9,10,11,12,13,14,15);
}
__device__ __forceinline__ v8f zero8() {
  v8f z;
#pragma unroll
  for (int i = 0; i < 8; ++i) z[i] = 0.0f;
  return z;
}
// D = A(16x32 bf16) * B(32x16 bf16) + C(16x16 f32)
__device__ __forceinline__ v8f wmma_bf16(v16bf a, v16bf b, v8f c) {
  return __builtin_amdgcn_wmma_f32_16x16x32_bf16(false, a, false, b,
                                                 (short)0, c, false, false);
}

// ---------------------------------------------------------------------------
// Tensor Data Mover: 2D tile (rows x row_elems of bf16) global -> LDS.
// Descriptor packing per CDNA5 ISA ch.8 (D# group0/group1); groups 2/3 (and
// the trailing group of the 6-arg builtin form) zero for a 2D tensor.
// flags_w0 carries data_size + LDS pad config.
// data_size=1 (2 bytes).  pad_interval code 5 => pad after every 256B row.
// ---------------------------------------------------------------------------
__device__ __forceinline__ void tdm_load_2d(unsigned lds_off, const void* gptr,
                                            unsigned tensor_d0, unsigned tensor_d1,
                                            unsigned tile_d0, unsigned tile_d1,
                                            unsigned stride_elems, unsigned flags_w0)
{
  unsigned long long ga = (unsigned long long)gptr;
  u32x4 g0;
  g0[0] = 1u;                                             // count=1 (valid D#)
  g0[1] = lds_off;                                        // lds_addr (bytes)
  g0[2] = (unsigned)(ga & 0xffffffffu);                   // global_addr[31:0]
  g0[3] = (unsigned)((ga >> 32) & 0x01ffffffu) | (2u << 30); // addr[56:32], type=2
  i32x8 g1;
  g1[0] = (int)flags_w0;                                  // mask=0|data_size|pad cfg
  g1[1] = (int)((tensor_d0 & 0xffffu) << 16);             // dim0[15:0] @ bits63:48
  g1[2] = (int)(((tensor_d0 >> 16) & 0xffffu) |
                ((tensor_d1 & 0xffffu) << 16));           // dim0[31:16] | dim1[15:0]
  g1[3] = (int)(((tensor_d1 >> 16) & 0xffffu) |
                ((tile_d0 & 0xffffu) << 16));             // dim1[31:16] | tile_dim0
  g1[4] = (int)(tile_d1 & 0xffffu);                       // tile_dim1 (tile_dim2=0)
  g1[5] = (int)stride_elems;                              // dim0_stride[31:0]
  g1[6] = 0;                                              // stride hi / dim1_stride
  g1[7] = 0;
  i32x4 z4;
  z4[0] = 0; z4[1] = 0; z4[2] = 0; z4[3] = 0;
  i32x8 z8;
#pragma unroll
  for (int i = 0; i < 8; ++i) z8[i] = 0;
  // amdgpu-toolchain (clang-23) 6-arg form
  __builtin_amdgcn_tensor_load_to_lds(g0, g1, z4, z4, z8, 0);
}

// flags: data_size=2B (bits17:16=1), pad_enable(bit20), pad_interval=5 (64 DW),
// pad_amount code (bits31:25) = dwords-1
#define TDM_FLAGS(padDwMinus1) \
  ((1u << 16) | (1u << 20) | (5u << 22) | ((unsigned)(padDwMinus1) << 25))

// ---------------------------------------------------------------------------
// GEMM: C[M,N] = A[M,K] @ B[K,N]   (fp32 in/out, bf16x3 split on WMMA)
// Tiles: BM=128, BN=128, BK=32.  256 threads = 8 waves, wave grid 2x4,
// each wave owns a 64x32 region = 4x2 WMMA tiles.
// LDS rows padded to 56 elems (112B = 7*16B): 16B-aligned b128 reads and
// conflict-free lane striding (gcd(28,64)=4 -> 16 distinct banks).
// ---------------------------------------------------------------------------
#define BM 128
#define BN 128
#define BK 32
#define LDP 56

__global__ __launch_bounds__(256)
void gemm_bf16x3_wmma(const float* __restrict__ A, const float* __restrict__ B,
                      float* __restrict__ C, int M, int N, int K)
{
  __shared__ __align__(16) __bf16 aH[BM][LDP];   // [m][k] hi
  __shared__ __align__(16) __bf16 aL[BM][LDP];   // [m][k] lo (residual)
  __shared__ __align__(16) __bf16 bH[BN][LDP];   // [n][k] hi (transposed)
  __shared__ __align__(16) __bf16 bL[BN][LDP];   // [n][k] lo

  const int tid  = threadIdx.x;
  const int lane = tid & 31;
  const int wave = tid >> 5;
  const int lm   = lane & 15;      // 0..15
  const int lg   = lane >> 4;      // 0..1 (lane group)
  const int wr   = wave >> 2;      // 0..1 -> 64-row group
  const int wc   = wave & 3;       // 0..3 -> 32-col group
  const int tm   = blockIdx.y;
  const int tn   = blockIdx.x;

  v8f acc[4][2];
#pragma unroll
  for (int rt = 0; rt < 4; ++rt)
#pragma unroll
    for (int ct = 0; ct < 2; ++ct) acc[rt][ct] = zero8();

  for (int k0 = 0; k0 < K; k0 += BK) {
    // ---- cooperative stage of A tile (128x32): 16 floats per thread ----
    {
      const int row = tid >> 1;
      const int ks  = (tid & 1) * 16;
      const float4* src = (const float4*)(A + (size_t)(tm * BM + row) * K + k0 + ks);
#pragma unroll
      for (int jj = 0; jj < 4; ++jj) {
        float4 v = src[jj];
        float e[4] = {v.x, v.y, v.z, v.w};
#pragma unroll
        for (int t = 0; t < 4; ++t) {
          __bf16 hx = (__bf16)e[t];
          aH[row][ks + jj * 4 + t] = hx;
          aL[row][ks + jj * 4 + t] = (__bf16)(e[t] - (float)hx);
        }
      }
    }
    // ---- cooperative stage of B tile (32x128) transposed into [n][k] ----
    {
      const int row = tid >> 3;          // k within tile: 0..31
      const int cs  = (tid & 7) * 16;    // n segment
      const float4* src = (const float4*)(B + (size_t)(k0 + row) * N + tn * BN + cs);
#pragma unroll
      for (int jj = 0; jj < 4; ++jj) {
        float4 v = src[jj];
        float e[4] = {v.x, v.y, v.z, v.w};
#pragma unroll
        for (int t = 0; t < 4; ++t) {
          __bf16 hx = (__bf16)e[t];
          bH[cs + jj * 4 + t][row] = hx;
          bL[cs + jj * 4 + t][row] = (__bf16)(e[t] - (float)hx);
        }
      }
    }
    if (k0 + BK < K) {  // hint next tiles toward L2 (global_prefetch_b8)
      __builtin_prefetch(A + (size_t)(tm * BM + (tid >> 1)) * K + k0 + BK + (tid & 1) * 16, 0, 0);
      __builtin_prefetch(B + (size_t)(k0 + BK + (tid >> 3)) * N + tn * BN + (tid & 7) * 16, 0, 0);
    }
    __syncthreads();

    // ---- fragments ----
    // A 16x32 bf16 layout: lanes 0-15 hold K 0..7 & 16..23; lanes 16-31 hold
    // K 8..15 & 24..31 (row m = lane%16).
    const int c0 = lg * 8;
    const int c1 = 16 + lg * 8;
    v16bf afH[4], afL[4], bfH[2], bfL[2];
#pragma unroll
    for (int rt = 0; rt < 4; ++rt) {
      const int r = wr * 64 + rt * 16 + lm;
      afH[rt] = cat16(ld8(&aH[r][c0]), ld8(&aH[r][c1]));
      afL[rt] = cat16(ld8(&aL[r][c0]), ld8(&aL[r][c1]));
    }
    // B 32x16 bf16 layout: column n = lane%16; lanes 0-15 hold K 0..7 & 16..23,
    // lanes 16-31 hold K 8..15 & 24..31.
#pragma unroll
    for (int ct = 0; ct < 2; ++ct) {
      const int c = wc * 32 + ct * 16 + lm;
      bfH[ct] = cat16(ld8(&bH[c][lg * 8]), ld8(&bH[c][16 + lg * 8]));
      bfL[ct] = cat16(ld8(&bL[c][lg * 8]), ld8(&bL[c][16 + lg * 8]));
    }
    // bf16x3: hi*hi + hi*lo + lo*hi (drop lo*lo) -> near-fp32 product
#pragma unroll
    for (int rt = 0; rt < 4; ++rt)
#pragma unroll
      for (int ct = 0; ct < 2; ++ct) {
        acc[rt][ct] = wmma_bf16(afH[rt], bfH[ct], acc[rt][ct]);
        acc[rt][ct] = wmma_bf16(afH[rt], bfL[ct], acc[rt][ct]);
        acc[rt][ct] = wmma_bf16(afL[rt], bfH[ct], acc[rt][ct]);
      }
    __syncthreads();
  }

  // C layout: VGPR i, lanes 0-15 -> (m=i, n=lane); lanes 16-31 -> (m=i+8).
#pragma unroll
  for (int rt = 0; rt < 4; ++rt)
#pragma unroll
    for (int ct = 0; ct < 2; ++ct) {
      const int rbase = tm * BM + wr * 64 + rt * 16 + 8 * lg;
      const int cidx  = tn * BN + wc * 32 + ct * 16 + lm;
#pragma unroll
      for (int i = 0; i < 8; ++i)
        C[(size_t)(rbase + i) * N + cidx] = acc[rt][ct][i];
    }
}

// ---------------------------------------------------------------------------
// RoPE + split qkv[4096][6144] f32 -> q,k,v bf16 in [b][h][s][d] layout.
// One block per (b,s,h); 64 threads; thread i handles rotation pair (i, i+64).
// ---------------------------------------------------------------------------
__global__ __launch_bounds__(64)
void rope_split_kernel(const float* __restrict__ qkv,
                       __bf16* __restrict__ qo, __bf16* __restrict__ ko,
                       __bf16* __restrict__ vo)
{
  const int idx = blockIdx.x;
  const int h = idx % N_HEADS;
  const int s = (idx / N_HEADS) % SEQ;
  const int b = idx / (N_HEADS * SEQ);
  const int i = threadIdx.x;                 // 0..63

  const size_t row   = (size_t)b * SEQ + s;
  const size_t qbase = row * (size_t)(3 * HID) + (size_t)h * HEAD_DIM;
  const size_t kbase = qbase + HID;
  const size_t vbase = qbase + 2 * HID;

  const float inv = __expf(-((float)(2 * i) / (float)HEAD_DIM) * logf(10000.0f));
  const float ang = (float)s * inv;
  const float c = cosf(ang), sn = sinf(ang);

  const float q1 = qkv[qbase + i],      q2 = qkv[qbase + 64 + i];
  const float k1 = qkv[kbase + i],      k2 = qkv[kbase + 64 + i];
  const float v1 = qkv[vbase + i],      v2 = qkv[vbase + 64 + i];

  const size_t ob = (((size_t)b * N_HEADS + h) * SEQ + s) * HEAD_DIM;
  qo[ob + i]      = (__bf16)(q1 * c - q2 * sn);
  qo[ob + 64 + i] = (__bf16)(q2 * c + q1 * sn);
  ko[ob + i]      = (__bf16)(k1 * c - k2 * sn);
  ko[ob + 64 + i] = (__bf16)(k2 * c + k1 * sn);
  vo[ob + i]      = (__bf16)v1;
  vo[ob + 64 + i] = (__bf16)v2;
}

// ---------------------------------------------------------------------------
// Flash attention with causal mask + T5 relative bias.
// Block = 128 threads (4 waves), each wave owns 16 query rows (QB=64 rows/WG).
// K/V 32-key tiles are DMA'd by the Tensor Data Mover into double-buffered
// LDS (wave 0 issues; s_wait_tensorcnt + barrier publishes), pipelined one
// tile ahead of the WMMA compute.
// K tile rows padded to 136 elems (272B, TDM pad 16B): b128 reads hit
// distinct banks.  V tile rows padded to 130 elems (260B, TDM pad 4B):
// the strided 16-bit gather for PV B-fragments steps banks by 1 per row.
// ---------------------------------------------------------------------------
#define QB   64
#define KBLK 32
#define KTP  136   // 128 + 8  (16B pad)
#define VTP  130   // 128 + 2  (4B pad)

__global__ __launch_bounds__(128)
void flash_attn_wmma(const __bf16* __restrict__ Q, const __bf16* __restrict__ Kv,
                     const __bf16* __restrict__ V, const float* __restrict__ table,
                     float* __restrict__ ctx)
{
  __shared__ __align__(16) __bf16 kt[2][KBLK][KTP];       // [buf][key][dim]
  __shared__ __align__(16) __bf16 vraw[2][KBLK][VTP];     // [buf][key][dim]
  __shared__ __align__(16) __bf16 pstage[4][16][LDP];     // per-wave P tile
  __shared__ float bias_s[SEQ];                            // bias per distance

  const int tid  = threadIdx.x;
  const int lane = tid & 31;
  const int w    = tid >> 5;
  const int lm   = lane & 15;
  const int lg   = lane >> 4;
  const int h  = blockIdx.y;
  const int b  = blockIdx.z;
  const int q0 = blockIdx.x * QB;
  const int qr0 = q0 + w * 16;

  // --- precompute bias[distance] = table[bucket(distance)][h] once ---
  for (int d = tid; d < SEQ; d += 128) {
    int bk;
    if (d < 16) {
      bk = d;                                   // exact buckets
    } else {
      // 16 + floor(log(d/16)/log(128/16) * 16), clipped to 31
      bk = 16 + (int)(logf((float)d * (1.0f / 16.0f)) * (16.0f / logf(8.0f)));
      if (bk > NUM_BUCKETS - 1) bk = NUM_BUCKETS - 1;
    }
    bias_s[d] = table[bk * N_HEADS + h];
  }

  const size_t headbase = ((size_t)b * N_HEADS + h) * SEQ;
  const unsigned kt_off[2] = {(unsigned)(size_t)&kt[0][0][0],
                              (unsigned)(size_t)&kt[1][0][0]};
  const unsigned vr_off[2] = {(unsigned)(size_t)&vraw[0][0][0],
                              (unsigned)(size_t)&vraw[1][0][0]};

  // --- resident Q fragments: 4 chunks of 32 dims ---
  v16bf qf[4];
#pragma unroll
  for (int dc = 0; dc < 4; ++dc) {
    const __bf16* qp = Q + (headbase + qr0 + lm) * HEAD_DIM + dc * 32;
    qf[dc] = cat16(ld8(qp + lg * 8), ld8(qp + 16 + lg * 8));
  }

  v8f o[8];
  float mrow[8], lrow[8];
#pragma unroll
  for (int t = 0; t < 8; ++t) o[t] = zero8();
#pragma unroll
  for (int i = 0; i < 8; ++i) { mrow[i] = -1e30f; lrow[i] = 0.0f; }

  const float scale = 0.08838834764831845f;   // 1/sqrt(128)
  const int kend = q0 + QB - 1;               // causal bound for this block

  // --- pipeline prologue: DMA first K/V tile into buffer 0 ---
  if (w == 0) {
    tdm_load_2d(kt_off[0], Kv + (headbase + 0) * HEAD_DIM,
                HEAD_DIM, SEQ, HEAD_DIM, KBLK, HEAD_DIM, TDM_FLAGS(3));
    tdm_load_2d(vr_off[0], V + (headbase + 0) * HEAD_DIM,
                HEAD_DIM, SEQ, HEAD_DIM, KBLK, HEAD_DIM, TDM_FLAGS(0));
  }

  for (int kb = 0; kb <= kend; kb += KBLK) {
    const int p = (kb >> 5) & 1;
    if (w == 0) __builtin_amdgcn_s_wait_tensorcnt(0);   // buffer p resident
    __syncthreads();                                    // publish to all waves
    // issue DMA for next tile into the other buffer (overlaps compute)
    if (w == 0 && kb + KBLK <= kend) {
      tdm_load_2d(kt_off[p ^ 1], Kv + (headbase + kb + KBLK) * HEAD_DIM,
                  HEAD_DIM, SEQ, HEAD_DIM, KBLK, HEAD_DIM, TDM_FLAGS(3));
      tdm_load_2d(vr_off[p ^ 1], V + (headbase + kb + KBLK) * HEAD_DIM,
                  HEAD_DIM, SEQ, HEAD_DIM, KBLK, HEAD_DIM, TDM_FLAGS(0));
    }

    // ---- S = Q @ K^T : two 16x16 score tiles over D=128 (4 k-chunks) ----
    v8f s0 = zero8(), s1 = zero8();
#pragma unroll
    for (int dc = 0; dc < 4; ++dc) {
      v16bf b0 = cat16(ld8(&kt[p][lm][dc * 32 + lg * 8]),
                       ld8(&kt[p][lm][dc * 32 + 16 + lg * 8]));
      v16bf b1 = cat16(ld8(&kt[p][16 + lm][dc * 32 + lg * 8]),
                       ld8(&kt[p][16 + lm][dc * 32 + 16 + lg * 8]));
      s0 = wmma_bf16(qf[dc], b0, s0);
      s1 = wmma_bf16(qf[dc], b1, s1);
    }

    // ---- online softmax (rows: m = i + 8*lg, cols spread over 16 lanes) ----
    float p0[8], p1[8], fsc[8];
#pragma unroll
    for (int i = 0; i < 8; ++i) {
      const int qrow = qr0 + i + 8 * lg;
      const int key0 = kb + lm;
      const int key1 = kb + 16 + lm;
      float t0 = (key0 <= qrow) ? (s0[i] * scale + bias_s[qrow - key0]) : -1e30f;
      float t1 = (key1 <= qrow) ? (s1[i] * scale + bias_s[qrow - key1]) : -1e30f;
      float mx = fmaxf(t0, t1);
      mx = fmaxf(mx, __shfl_xor(mx, 1));
      mx = fmaxf(mx, __shfl_xor(mx, 2));
      mx = fmaxf(mx, __shfl_xor(mx, 4));
      mx = fmaxf(mx, __shfl_xor(mx, 8));
      const float mn = fmaxf(mrow[i], mx);
      const float f  = __expf(mrow[i] - mn);
      const float e0 = __expf(t0 - mn);
      const float e1 = __expf(t1 - mn);
      float ps = e0 + e1;
      ps += __shfl_xor(ps, 1);
      ps += __shfl_xor(ps, 2);
      ps += __shfl_xor(ps, 4);
      ps += __shfl_xor(ps, 8);
      lrow[i] = lrow[i] * f + ps;
      mrow[i] = mn;
      fsc[i] = f; p0[i] = e0; p1[i] = e1;
    }

    // ---- rescale running context accumulators ----
#pragma unroll
    for (int t = 0; t < 8; ++t)
#pragma unroll
      for (int i = 0; i < 8; ++i) o[t][i] *= fsc[i];

    // ---- stage P (C-layout -> LDS) and reload as A-fragment ----
#pragma unroll
    for (int i = 0; i < 8; ++i) {
      pstage[w][i + 8 * lg][lm]      = (__bf16)p0[i];
      pstage[w][i + 8 * lg][16 + lm] = (__bf16)p1[i];
    }
    v16bf pf = cat16(ld8(&pstage[w][lm][8 * lg]),
                     ld8(&pstage[w][lm][16 + 8 * lg]));

    // ---- O += P @ V : 8 dim-tiles of 16; B-fragment gathered from the
    //      padded raw V tile (column = dim, K = key) ----
#pragma unroll
    for (int dt = 0; dt < 8; ++dt) {
      v16bf vf;
#pragma unroll
      for (int j = 0; j < 16; ++j)
        vf[j] = vraw[p][lg * 8 + j + (j >= 8 ? 8 : 0)][dt * 16 + lm];
      o[dt] = wmma_bf16(pf, vf, o[dt]);
    }
    __syncthreads();   // all waves done with buffer p before it is re-filled
  }

  // ---- normalize and store ctx as f32 rows [b*S + s][h*128 + d] ----
#pragma unroll
  for (int i = 0; i < 8; ++i) lrow[i] = 1.0f / lrow[i];
#pragma unroll
  for (int dt = 0; dt < 8; ++dt)
#pragma unroll
    for (int i = 0; i < 8; ++i) {
      const size_t r = (size_t)b * SEQ + qr0 + i + 8 * lg;
      ctx[r * HID + h * HEAD_DIM + dt * 16 + lm] = o[dt][i] * lrow[i];
    }
}

// ---------------------------------------------------------------------------
// Launch: qkv-gemm -> rope/split -> flash-attn -> dense-gemm
// Workspace: qkv f32 (100.7MB) | q bf16 | k bf16 | v bf16 (16.8MB each)
// ctx f32 aliases the qkv region (dead after rope).  Total ~151 MB.
// ---------------------------------------------------------------------------
extern "C" void kernel_launch(void* const* d_in, const int* in_sizes, int n_in,
                              void* d_out, int out_size, void* d_ws, size_t ws_size,
                              hipStream_t stream)
{
  (void)in_sizes; (void)n_in; (void)out_size; (void)ws_size;
  const float* x       = (const float*)d_in[0];
  const float* qkv_w   = (const float*)d_in[1];
  const float* dense_w = (const float*)d_in[2];
  const float* table   = (const float*)d_in[3];
  float* out = (float*)d_out;

  const size_t M = (size_t)BATCH * SEQ;                 // 4096
  float* qkv = (float*)d_ws;                            // M x 6144 f32
  size_t off = M * (size_t)(3 * HID) * sizeof(float);
  const size_t qkv_elems = (size_t)BATCH * N_HEADS * SEQ * HEAD_DIM;
  __bf16* qb   = (__bf16*)((char*)d_ws + off);
  __bf16* kbuf = qb + qkv_elems;
  __bf16* vbuf = kbuf + qkv_elems;
  float* ctx = qkv;                                     // alias (qkv dead)

  gemm_bf16x3_wmma<<<dim3((3 * HID) / BN, M / BM), 256, 0, stream>>>(
      x, qkv_w, qkv, (int)M, 3 * HID, HID);

  rope_split_kernel<<<BATCH * SEQ * N_HEADS, 64, 0, stream>>>(qkv, qb, kbuf, vbuf);

  flash_attn_wmma<<<dim3(SEQ / QB, N_HEADS, BATCH), 128, 0, stream>>>(
      qb, kbuf, vbuf, table, ctx);

  gemm_bf16x3_wmma<<<dim3(HID / BN, M / BM), 256, 0, stream>>>(
      ctx, dense_w, out, (int)M, HID, HID);
}
